// MultiHeadAttention_28965259444295
// MI455X (gfx1250) — matrix operand order repaired and checked by reading, not compiled, not run on hardware
//
#include <hip/hip_runtime.h>
#include <hip/hip_bf16.h>
#include <math.h>

// ---------------------------------------------------------------------------
// Local-window multi-head attention for MI455X (gfx1250, wave32, WMMA).
//   B=2, S=4096, d_model=1024, H=16, d_k=64, window=128.
// bf16 inputs / fp32 accumulation via v_wmma_f32_16x16x32_bf16.
// GEMMs: B-tile staged in LDS via async global->LDS copies (double buffered),
// A fragments software-pipelined in VGPRs.
// ---------------------------------------------------------------------------

typedef __bf16 bf16_t;
typedef __attribute__((ext_vector_type(16))) __bf16 v16bf;
typedef __attribute__((ext_vector_type(8)))  __bf16 v8bf;
typedef __attribute__((ext_vector_type(4)))  __bf16 v4bf;
typedef __attribute__((ext_vector_type(8)))  float  v8f;

// GCC-style int4 vector: the async-to-LDS builtin's pointee type per clang.
typedef int v4i __attribute__((vector_size(16)));
typedef __attribute__((address_space(1))) v4i* gvec_p;   // global
typedef __attribute__((address_space(3))) v4i* lvec_p;   // LDS

constexpr int MD    = 1024;   // d_model
constexpr int NH    = 16;     // heads
constexpr int DK    = 64;     // head dim
constexpr int SEQL  = 4096;   // sequence
constexpr int BB    = 2;      // batch
constexpr int WIN   = 128;    // local window
constexpr int MROWS = BB * SEQL;          // 8192 rows of x
constexpr float SCALEF = 0.125f;          // 1/sqrt(64)

constexpr int BPITCH = 40;    // LDS B-tile row pitch in halves (bank-conflict pad)

#if __has_builtin(__builtin_amdgcn_global_load_async_to_lds_b128)
#define HAVE_ASYNC_LDS 1
#else
#define HAVE_ASYNC_LDS 0
#endif

__device__ __forceinline__ void wait_async0() {
#if __has_builtin(__builtin_amdgcn_s_wait_asynccnt)
  __builtin_amdgcn_s_wait_asynccnt(0);
#else
  asm volatile("s_wait_asynccnt 0x0" ::: "memory");
#endif
}

union V16U { v16bf v; v8bf h[2]; };

// A-fragment (bf16, 16x32, M x K): rows of `base` are M, row pitch `ld` halves.
// lane<16 -> elems 0..15 = K k0..k0+15;
//             lane>=16 -> K k0+16..k0+31 (matches B-fragment order).
__device__ __forceinline__ v16bf load_a16(const bf16_t* base, int ld, int k0, int lane) {
  const bf16_t* rp = base + (size_t)(lane & 15) * ld + k0 + ((lane & 16) ? 16 : 0);
  V16U u;
  u.h[0] = *(const v8bf*)(rp);
  u.h[1] = *(const v8bf*)(rp + 8);
  return u.v;
}

// B-fragment (bf16, 32x16, K x N) where B[k][n] = rows[n][k] (rows row-major).
// Lane<16: column n=lane, K k0..k0+15; lane>=16: same column, K k0+16..k0+31.
__device__ __forceinline__ v16bf load_b16(const bf16_t* rows, int ld, int k0, int lane) {
  const bf16_t* p = rows + (size_t)(lane & 15) * ld + k0 + ((lane & 16) ? 16 : 0);
  V16U u;
  u.h[0] = *(const v8bf*)(p);
  u.h[1] = *(const v8bf*)(p + 8);
  return u.v;
}

// B-fragment from an LDS tile Bt[64][BPITCH] (n-major), sub-tile t (cols 16t..16t+15)
__device__ __forceinline__ v16bf load_b_lds(const bf16_t* Bt, int t, int lane) {
  const bf16_t* p = Bt + (size_t)(t * 16 + (lane & 15)) * BPITCH + ((lane & 16) ? 16 : 0);
  V16U u;
  u.h[0] = *(const v8bf*)(p);
  u.h[1] = *(const v8bf*)(p + 8);
  return u.v;
}

// ---------------------------------------------------------------------------
// f32 -> bf16 conversion, 4 elements / thread (all sizes are multiples of 1024)
// ---------------------------------------------------------------------------
__global__ __launch_bounds__(256)
void cvt_f32_bf16(const float* __restrict__ in, bf16_t* __restrict__ out, int n) {
  int i = (blockIdx.x * 256 + threadIdx.x) * 4;
  if (i + 3 < n) {
    float4 f = *(const float4*)(in + i);
    v4bf o;
    o[0] = (bf16_t)f.x; o[1] = (bf16_t)f.y; o[2] = (bf16_t)f.z; o[3] = (bf16_t)f.w;
    *(v4bf*)(out + i) = o;
  }
}

// ---------------------------------------------------------------------------
// GEMM: out = A @ W^T + bias.  A: [M,K] bf16 row-major, W: [N,K] bf16 row-major.
// Block = 256 threads = 8 waves stacked along M; block tile = 256 rows x 64 cols.
// Wave tile = 32 x 64 (2 M-subtiles x 4 N-subtiles = 8 accumulators).
// B-tile (64 x 32) staged in LDS via async copies, double buffered and shared by
// all 8 waves; A fragments double-buffered in VGPRs (software pipeline).
// mode 0: f32 out [M,N] row-major          (final O projection -> d_out)
// mode 1: bf16 out [B,H,S,DK]              (Q, K)
// mode 2: bf16 out [B,H,DK,S] (transposed) (V, for contiguous P@V B-frags)
// ---------------------------------------------------------------------------
__global__ __launch_bounds__(256)
void gemm_xwT(const bf16_t* __restrict__ A, const bf16_t* __restrict__ W,
              const float* __restrict__ bias, void* __restrict__ out,
              int M, int N, int K, int mode) {
  __shared__ __align__(16) bf16_t Bt[2][64][BPITCH];

  const int tid   = threadIdx.x;
  const int lane  = tid & 31;
  const int wv    = tid >> 5;
  const int nBase = blockIdx.x * 64;
  const int mBase = blockIdx.y * 256 + wv * 32;

  // Cooperative B staging: thread tid moves 16B: row = tid/4, k-part = (tid%4)*8
  const int srow = tid >> 2;
  const int skp  = (tid & 3) * 8;
  const bf16_t* sgp_base = W + (size_t)(nBase + srow) * K + skp;

  auto stage_b = [&](int buf, int k0) {
    const bf16_t* gp = sgp_base + k0;
    bf16_t* lp = &Bt[buf][srow][skp];
#if HAVE_ASYNC_LDS
    __builtin_amdgcn_global_load_async_to_lds_b128((gvec_p)gp, (lvec_p)lp, 0, 0);
#else
    *(v8bf*)lp = *(const v8bf*)gp;
#endif
  };

  v8f acc[2][4] = {};
  const bf16_t* A0 = A + (size_t)mBase * K;        // rows mBase .. +15
  const bf16_t* A1 = A + (size_t)(mBase + 16) * K; // rows mBase+16 .. +31

  // Prologue: stage first B-tile, preload first A fragments.
  stage_b(0, 0);
  v16bf a0c = load_a16(A0, K, 0, lane);
  v16bf a1c = load_a16(A1, K, 0, lane);
#if HAVE_ASYNC_LDS
  wait_async0();
#endif
  __syncthreads();

  for (int k0 = 0; k0 < K; k0 += 32) {
    const int cur = (k0 >> 5) & 1;
    const bool more = (k0 + 32) < K;

    v16bf a0n, a1n;
    if (more) {
      stage_b(cur ^ 1, k0 + 32);             // async issue, overlaps compute
      a0n = load_a16(A0, K, k0 + 32, lane);  // pipelined A fragments
      a1n = load_a16(A1, K, k0 + 32, lane);
    }

#pragma unroll
    for (int t = 0; t < 4; ++t) {
      v16bf bfb = load_b_lds(&Bt[cur][0][0], t, lane);
      acc[0][t] = __builtin_amdgcn_wmma_f32_16x16x32_bf16(
          false, a0c, false, bfb, (short)0, acc[0][t], false, false);
      acc[1][t] = __builtin_amdgcn_wmma_f32_16x16x32_bf16(
          false, a1c, false, bfb, (short)0, acc[1][t], false, false);
    }

    a0c = a0n; a1c = a1n;
    if (more) {
#if HAVE_ASYNC_LDS
      wait_async0();                         // this wave's stage of buf cur^1 done
#endif
      __syncthreads();                       // all waves staged + done reading cur
    }
  }

  const int hi  = lane >> 4;   // C layout: VGPR r, hi half -> M = r + 8*hi
  const int l15 = lane & 15;   // N = lane & 15
#pragma unroll
  for (int mm = 0; mm < 2; ++mm) {
#pragma unroll
    for (int t = 0; t < 4; ++t) {
      int n = nBase + 16 * t + l15;
      float bv = bias[n];
#pragma unroll
      for (int r = 0; r < 8; ++r) {
        float val = acc[mm][t][r] + bv;
        int gm = mBase + mm * 16 + hi * 8 + r;
        if (mode == 0) {
          ((float*)out)[(size_t)gm * N + n] = val;
        } else {
          int b = gm >> 12;                 // / SEQL
          int s = gm & (SEQL - 1);
          int h = n >> 6;                   // / DK
          int d = n & (DK - 1);
          bf16_t bw = (bf16_t)val;
          if (mode == 1)
            ((bf16_t*)out)[(((size_t)b * NH + h) * SEQL + s) * DK + d] = bw;
          else
            ((bf16_t*)out)[(((size_t)b * NH + h) * DK + d) * SEQL + s] = bw;
        }
      }
    }
  }
}

// ---------------------------------------------------------------------------
// Local attention. One wave per 16-query tile; 4 waves / block.
//   q,k: [B,H,S,64] bf16.  vt: [B,H,64,S] bf16 (transposed V).
//   ctx: [B,S,1024] bf16.
// Keys per tile: [max(0,qbase-128) .. qbase+15] -> <= 144 keys = 9 key tiles.
// ---------------------------------------------------------------------------
__global__ __launch_bounds__(128)
void attn_local(const bf16_t* __restrict__ qb, const bf16_t* __restrict__ kb,
                const bf16_t* __restrict__ vt, bf16_t* __restrict__ ctx) {
  __shared__ __align__(16) float  sc[4][16][160];   // raw scores / exps
  __shared__ __align__(16) bf16_t pp[4][16][160];   // softmax probs (bf16)

  const int lane = threadIdx.x & 31;
  const int wv   = threadIdx.x >> 5;
  const int hi   = lane >> 4;
  const int l15  = lane & 15;

  const int tile  = blockIdx.x * 4 + wv;
  const int qt    = tile & (SEQL / 16 - 1);   // 256 tiles per (b,h)
  const int h     = (tile >> 8) & (NH - 1);
  const int b     = tile >> 12;
  const int qbase = qt * 16;

  const int kstart = (qbase >= WIN) ? (qbase - WIN) : 0;
  const int nk     = qbase + 16 - kstart;     // <= 144
  const int nkt    = nk >> 4;                 // <= 9 key tiles

  const bf16_t* Q  = qb + (((size_t)b * NH + h) * SEQL + qbase) * DK;
  const bf16_t* Kp = kb + (((size_t)b * NH + h) * SEQL) * DK;
  const bf16_t* Vt = vt + (((size_t)b * NH + h) * DK) * SEQL;

  // Q fragments for K-dim 0..31 and 32..63
  v16bf aq0 = load_a16(Q, DK, 0, lane);
  v16bf aq1 = load_a16(Q, DK, 32, lane);

  // ---- scores = (Q K^T) * scale, masked, into LDS ----
  for (int kt = 0; kt < nkt; ++kt) {
    const bf16_t* Krow = Kp + (size_t)(kstart + kt * 16) * DK;
    v16bf b0 = load_b16(Krow, DK, 0, lane);
    v16bf b1 = load_b16(Krow, DK, 32, lane);
    v8f acc = {};
    acc = __builtin_amdgcn_wmma_f32_16x16x32_bf16(false, aq0, false, b0, (short)0, acc, false, false);
    acc = __builtin_amdgcn_wmma_f32_16x16x32_bf16(false, aq1, false, b1, (short)0, acc, false, false);
#pragma unroll
    for (int r = 0; r < 8; ++r) {
      int m  = r + hi * 8;
      int qi = qbase + m;
      int kj = kstart + kt * 16 + l15;
      float v = acc[r] * SCALEF;
      if (kj > qi || kj + WIN < qi) v = -10000.0f;   // reference mask fill
      sc[wv][m][kt * 16 + l15] = v;
    }
  }
  asm volatile("s_wait_dscnt 0x0" ::: "memory");
  __syncthreads();

  // ---- per-row softmax: lane l15 owns row l15; each half-wave scans half the
  //      columns, partials combined across the halves with shfl_xor(16). ----
  {
    const int m    = l15;
    const int half = nk >> 1;                 // nk is a multiple of 16
    const int c0   = hi ? half : 0;
    const int c1   = hi ? nk : half;
    float mx = -1e30f;
    for (int c = c0; c < c1; ++c) mx = fmaxf(mx, sc[wv][m][c]);
    mx = fmaxf(mx, __shfl_xor(mx, 16, 32));
    float sum = 0.f;
    for (int c = c0; c < c1; ++c) {
      float e = __expf(sc[wv][m][c] - mx);
      sc[wv][m][c] = e;
      sum += e;
    }
    sum += __shfl_xor(sum, 16, 32);
    float inv = 1.0f / sum;
    for (int c = c0; c < c1; ++c) pp[wv][m][c] = (bf16_t)(sc[wv][m][c] * inv + 1e-9f);
    // zero-pad K columns nk..159, split between the two halves
    const int pmid = nk + ((160 - nk) >> 1);
    const int p0   = hi ? pmid : nk;
    const int p1   = hi ? 160 : pmid;
    for (int c = p0; c < p1; ++c) pp[wv][m][c] = (bf16_t)0.0f;
  }
  asm volatile("s_wait_dscnt 0x0" ::: "memory");
  __syncthreads();

  // ---- ctx = P @ V  (A-frag from LDS, B-frag contiguous from transposed V) ----
  v8f cacc[4] = {};
  const int nks32 = (nk + 31) >> 5;   // <= 5 K-steps of 32
  const bf16_t* prow = &pp[wv][0][0];
  for (int kk = 0; kk < nks32; ++kk) {
    int k0 = kk * 32;
    v16bf ap = load_a16(prow, 160, k0, lane);
#pragma unroll
    for (int t = 0; t < 4; ++t) {
      int kof  = hi ? 16 : 0;
      int kbeg = kstart + k0 + kof;
      if (kbeg > SEQL - 16) kbeg = SEQL - 16;  // clamp; P is zero there anyway
      const bf16_t* p = Vt + (size_t)(16 * t + l15) * SEQL + kbeg;
      V16U u;
      u.h[0] = *(const v8bf*)(p);
      u.h[1] = *(const v8bf*)(p + 8);
      cacc[t] = __builtin_amdgcn_wmma_f32_16x16x32_bf16(
          false, ap, false, u.v, (short)0, cacc[t], false, false);
    }
  }

  // ---- store ctx tile: [B,S,MD] bf16, column = h*64 + d ----
  bf16_t* cbase = ctx + ((size_t)b * SEQL) * MD + (size_t)h * DK;
#pragma unroll
  for (int t = 0; t < 4; ++t) {
#pragma unroll
    for (int r = 0; r < 8; ++r) {
      int s = qbase + r + hi * 8;
      int d = 16 * t + l15;
      cbase[(size_t)s * MD + d] = (bf16_t)cacc[t][r];
    }
  }
}

// ---------------------------------------------------------------------------
// Host launcher. Inputs (setup_inputs order):
//   0:x  1:Wq 2:bq 3:Wk 4:bk 5:Wv 6:bv 7:Wo 8:bo     out: f32 [B,S,MD]
// Workspace: bf16 copies of x/weights + Q,K,Vt,ctx  (~88 MB).
// ---------------------------------------------------------------------------
extern "C" void kernel_launch(void* const* d_in, const int* in_sizes, int n_in,
                              void* d_out, int out_size, void* d_ws, size_t ws_size,
                              hipStream_t stream) {
  (void)in_sizes; (void)n_in; (void)out_size; (void)ws_size;
  const float* x  = (const float*)d_in[0];
  const float* Wq = (const float*)d_in[1];
  const float* bq = (const float*)d_in[2];
  const float* Wk = (const float*)d_in[3];
  const float* bk = (const float*)d_in[4];
  const float* Wv = (const float*)d_in[5];
  const float* bv = (const float*)d_in[6];
  const float* Wo = (const float*)d_in[7];
  const float* bo = (const float*)d_in[8];
  float* out = (float*)d_out;

  char* ws = (char*)d_ws;
  size_t off = 0;
  auto carve = [&](size_t bytes) -> void* {
    void* p = ws + off;
    off += (bytes + 255) & ~(size_t)255;
    return p;
  };
  bf16_t* xb   = (bf16_t*)carve((size_t)MROWS * MD * sizeof(bf16_t));
  bf16_t* wqb  = (bf16_t*)carve((size_t)MD * MD * sizeof(bf16_t));
  bf16_t* wkb  = (bf16_t*)carve((size_t)MD * MD * sizeof(bf16_t));
  bf16_t* wvb  = (bf16_t*)carve((size_t)MD * MD * sizeof(bf16_t));
  bf16_t* wob  = (bf16_t*)carve((size_t)MD * MD * sizeof(bf16_t));
  bf16_t* qbuf = (bf16_t*)carve((size_t)MROWS * MD * sizeof(bf16_t)); // [B,H,S,DK]
  bf16_t* kbuf = (bf16_t*)carve((size_t)MROWS * MD * sizeof(bf16_t)); // [B,H,S,DK]
  bf16_t* vtbf = (bf16_t*)carve((size_t)MROWS * MD * sizeof(bf16_t)); // [B,H,DK,S]
  bf16_t* ctxb = (bf16_t*)carve((size_t)MROWS * MD * sizeof(bf16_t)); // [B,S,MD]

  const int nx = MROWS * MD;   // 8388608
  const int nw = MD * MD;      // 1048576
  cvt_f32_bf16<<<nx / 1024, 256, 0, stream>>>(x,  xb,  nx);
  cvt_f32_bf16<<<nw / 1024, 256, 0, stream>>>(Wq, wqb, nw);
  cvt_f32_bf16<<<nw / 1024, 256, 0, stream>>>(Wk, wkb, nw);
  cvt_f32_bf16<<<nw / 1024, 256, 0, stream>>>(Wv, wvb, nw);
  cvt_f32_bf16<<<nw / 1024, 256, 0, stream>>>(Wo, wob, nw);

  dim3 gg(MD / 64, MROWS / 256);   // (16, 32) blocks of 256 threads
  gemm_xwT<<<gg, 256, 0, stream>>>(xb, wqb, bq, qbuf, MROWS, MD, MD, 1);
  gemm_xwT<<<gg, 256, 0, stream>>>(xb, wkb, bk, kbuf, MROWS, MD, MD, 1);
  gemm_xwT<<<gg, 256, 0, stream>>>(xb, wvb, bv, vtbf, MROWS, MD, MD, 2);

  const int ntiles = BB * NH * (SEQL / 16);  // 8192 q-tiles, 4 waves/block
  attn_local<<<ntiles / 4, 128, 0, stream>>>(qbuf, kbuf, vtbf, ctxb);

  gemm_xwT<<<gg, 256, 0, stream>>>(ctxb, wob, bo, out, MROWS, MD, MD, 0);
}
